// ImplicitGenerator3d_26164940767669
// MI455X (gfx1250) — matrix operand order repaired and checked by reading, not compiled, not run on hardware
//
#include <hip/hip_runtime.h>
#include <hip/hip_fp16.h>

typedef __attribute__((ext_vector_type(16))) _Float16 v16h;
typedef __attribute__((ext_vector_type(8)))  float    v8f;
typedef __attribute__((ext_vector_type(4)))  unsigned v4u;
typedef __attribute__((ext_vector_type(8)))  int      v8i;
typedef __attribute__((ext_vector_type(4)))  int      v4i;

#define HID     128
#define NLAYERS 6
#define ZDIM    256
#define IMG     64
#define SSTEPS  12
#define HW      (IMG*IMG)          // 4096 rays per batch
#define PPB     (HW*SSTEPS)        // 49152 points per batch per eval
#define BATCH   2
#define NPTS    (BATCH*PPB)        // 98304 points per eval
#define NRAYS   (BATCH*HW)         // 8192
#define FREQ_N  (NLAYERS*HID)      // 768

// ---------------------------------------------------------------- hash RNG
__device__ __forceinline__ float u01hash(unsigned x) {
  x ^= x >> 17; x *= 0xed5ad4bbu;
  x ^= x >> 11; x *= 0xac4c1b51u;
  x ^= x >> 15; x *= 0x31848babu;
  x ^= x >> 14;
  return (float)(x >> 8) * 5.9604644775390625e-08f;   // [0,1)
}

// ------------------------------------------------- TDM: 128x128 f16 tile -> LDS
// Descriptor per CDNA5 ISA. 2-D tensor, data_size=2B,
// tile 128x128, tensor_dim0_stride=128. Groups 2/3 zero (<=2D).
__device__ __forceinline__ void tdm_load_tile_128x128_f16(const _Float16* gsrc,
                                                          void* lds_dst) {
  const unsigned lds_off = (unsigned)(uintptr_t)lds_dst;         // LDS byte offset
  const unsigned long long ga = (unsigned long long)(uintptr_t)gsrc;
  v4u g0 = { 1u,                                   // count=1 (valid user D#)
             lds_off,                              // lds_addr  [63:32]
             (unsigned)(ga & 0xffffffffu),         // global_addr[95:64]
             (unsigned)((ga >> 32) & 0x01ffffffu)  // global_addr[120:96]
               | (2u << 30) };                     // type=2 ("image") [127:126]
  v8i g1 = { (int)(1u << 16),        // workgroup_mask=0, data_size=1 (2 bytes)
             (int)(128u << 16),      // tensor_dim0[15:0] at bits [63:48]
             (int)(128u << 16),      // tensor_dim0 hi=0 | tensor_dim1 lo at [95:80]
             (int)(128u << 16),      // tensor_dim1 hi=0 | tile_dim0 at [127:112]
             (int)(128u),            // tile_dim1 at [143:128], tile_dim2=0
             (int)(128u),            // tensor_dim0_stride[31:0]
             0,                      // stride0 hi | tensor_dim1_stride lo
             0 };
  v4i z4 = { 0, 0, 0, 0 };
#if defined(__clang_major__) && (__clang_major__ >= 23)
  v8i z8 = { 0, 0, 0, 0, 0, 0, 0, 0 };
  __builtin_amdgcn_tensor_load_to_lds(g0, g1, z4, z4, z8, 0);
#else
  __builtin_amdgcn_tensor_load_to_lds(g0, g1, z4, z4, 0);
#endif
}

// ------------------------------------------------------------ mapping net
__global__ void mapping_kernel(const float* __restrict__ z,
                               const float* __restrict__ w0, const float* __restrict__ b0,
                               const float* __restrict__ w1, const float* __restrict__ b1,
                               const float* __restrict__ w2, const float* __restrict__ b2,
                               float* __restrict__ freqs, float* __restrict__ phases) {
  __shared__ float s0[ZDIM];
  __shared__ float s1[ZDIM];
  const int b = blockIdx.x, t = threadIdx.x;
  s0[t] = z[b*ZDIM + t];
  __syncthreads();
  float acc = b0[t];
  for (int k = 0; k < ZDIM; ++k) acc += s0[k]*w0[k*ZDIM + t];
  acc = acc > 0.f ? acc : 0.2f*acc;
  s1[t] = acc;
  __syncthreads();
  acc = b1[t];
  for (int k = 0; k < ZDIM; ++k) acc += s1[k]*w1[k*ZDIM + t];
  acc = acc > 0.f ? acc : 0.2f*acc;
  __syncthreads();
  s0[t] = acc;                      // reuse s0 as h1
  __syncthreads();
  for (int j = 0; j < 6; ++j) {
    const int n = t + j*256;        // 1536 outputs
    float v = b2[n];
    for (int k = 0; k < ZDIM; ++k) v += s0[k]*w2[k*(2*FREQ_N) + n];
    if (n < FREQ_N) freqs [b*FREQ_N + n]            = v*15.f + 30.f;
    else            phases[b*FREQ_N + (n - FREQ_N)] = v;
  }
}

// ---------------------------------------------- hid_w f32[l][k][n] -> f16[l][n][k]
__global__ void wcvt_kernel(const float* __restrict__ hw, _Float16* __restrict__ dst) {
  const int i = blockIdx.x*blockDim.x + threadIdx.x;
  if (i >= (NLAYERS-1)*HID*HID) return;
  const int layer = i / (HID*HID);
  const int r = i % (HID*HID);
  const int k = r / HID, n = r % HID;
  dst[layer*HID*HID + n*HID + k] = (_Float16)hw[i];
}

// ------------------------------------------------------------- ray setup
__global__ void gen_coarse_kernel(const float* __restrict__ c2w,
                                  float* __restrict__ tpts, float* __restrict__ zc,
                                  float* __restrict__ tdirs, float* __restrict__ torig) {
  const int idx = blockIdx.x*blockDim.x + threadIdx.x;
  if (idx >= NPTS) return;
  const int b   = idx / PPB;
  const int rem = idx % PPB;
  const int pix = rem / SSTEPS;
  const int s   = rem % SSTEPS;
  const int py = pix / IMG, px = pix % IMG;
  const float x = -1.f + 2.f*(float)px/(float)(IMG-1);
  const float y =  1.f - 2.f*(float)py/(float)(IMG-1);
  const float zcam = -9.514364454222585f;             // -1/tan(12deg/2)
  const float inv = rsqrtf(x*x + y*y + zcam*zcam);
  const float dx = x*inv, dy = y*inv, dz = zcam*inv;
  const float spacing = (1.12f - 0.88f) / (float)(SSTEPS-1);
  const float u = u01hash((unsigned)idx*2654435761u + 12345u);
  const float zv = 0.88f + (float)s*spacing + (u - 0.5f)*spacing;
  const float p0 = dx*zv, p1 = dy*zv, p2 = dz*zv;
  const float* M = c2w + b*16;
  tpts[idx*3+0] = M[0]*p0 + M[1]*p1 + M[2]*p2  + M[3];
  tpts[idx*3+1] = M[4]*p0 + M[5]*p1 + M[6]*p2  + M[7];
  tpts[idx*3+2] = M[8]*p0 + M[9]*p1 + M[10]*p2 + M[11];
  zc[idx] = zv;
  if (s == 0) {
    const int r = b*HW + pix;
    tdirs[r*3+0] = M[0]*dx + M[1]*dy + M[2]*dz;
    tdirs[r*3+1] = M[4]*dx + M[5]*dy + M[6]*dz;
    tdirs[r*3+2] = M[8]*dx + M[9]*dy + M[10]*dz;
    torig[r*3+0] = M[3];  torig[r*3+1] = M[7];  torig[r*3+2] = M[11];
  }
}

// --------------------------------------------------------- SIREN (WMMA + TDM)
// 1 block = 128 points, 8 waves. Activations (32KB) + double-buffered weight
// tiles (2x32KB) in LDS; weights streamed by the Tensor Data Mover so layer
// L+1 DMA overlaps layer L's 32 WMMAs per wave.
__global__ void __launch_bounds__(256)
siren_kernel(const float* __restrict__ pts,
             const float* __restrict__ freqs, const float* __restrict__ phases,
             const float* __restrict__ first_w, const float* __restrict__ first_b,
             const _Float16* __restrict__ hw16, const float* __restrict__ hid_b,
             const float* __restrict__ out_w, const float* __restrict__ out_b,
             float* __restrict__ out) {
  __shared__ __align__(16) _Float16 sA[HID*HID];       // 32KB activations [m][k]
  __shared__ __align__(16) _Float16 sW[2][HID*HID];    // 64KB weights^T   [n][k]
  __shared__ __align__(16) float    sPts[HID*3];       // 1.5KB point staging
  const int tid  = threadIdx.x;
  const int lane = tid & 31, wave = tid >> 5;
  const int base = blockIdx.x * HID;                   // first point row
  const int b    = base / PPB;                         // whole block same batch
  const float* fq = freqs  + b*FREQ_N;
  const float* ph = phases + b*FREQ_N;

  // kick off weight DMA for layers 0 and 1 immediately (TENSORcnt = 2)
  if (wave == 0) {
    tdm_load_tile_128x128_f16(hw16 + 0*HID*HID, &sW[0][0]);
    tdm_load_tile_128x128_f16(hw16 + 1*HID*HID, &sW[1][0]);
  }

  // ---- stage 128 points, overlapped with the TDM loads
  for (int i = tid; i < HID*3; i += 256) sPts[i] = pts[(size_t)base*3 + i];
  __syncthreads();

  // ---- first layer: 3 -> 128, FiLM sin (layer-0 freq/phase slice)
  for (int j = 0; j < 64; ++j) {
    const int idx = j*256 + tid;
    const int m = idx >> 7, n = idx & 127;
    const float s = sPts[m*3+0]*first_w[n] + sPts[m*3+1]*first_w[HID+n]
                  + sPts[m*3+2]*first_w[2*HID+n] + first_b[n];
    sA[m*HID + n] = (_Float16)__sinf(fq[n]*s + ph[n]);
  }
  __syncthreads();

  const int m0   = wave * 16;
  const int mrow = m0 + (lane & 15);
  const int hi   = lane >> 4;

  union Frag { unsigned u[8]; v16h v; };

  for (int layer = 0; layer < NLAYERS-1; ++layer) {
    // gate on this layer's DMA (in-order completion: <=1 leaves L+1 in flight)
    if (wave == 0) {
      if (layer < NLAYERS-2) __builtin_amdgcn_s_wait_tensorcnt(1);
      else                   __builtin_amdgcn_s_wait_tensorcnt(0);
    }
    __syncthreads();
    const _Float16* sWl = sW[layer & 1];

    v8f acc[8];
    const v8f vzero = {0.f,0.f,0.f,0.f,0.f,0.f,0.f,0.f};
    #pragma unroll
    for (int nt = 0; nt < 8; ++nt) acc[nt] = vzero;

    // A fragments: 16x32 f16, lane half selects K 0-7/16-23 vs 8-15/24-31
    Frag af[4];
    #pragma unroll
    for (int kc = 0; kc < 4; ++kc) {
      const _Float16* arow = sA + mrow*HID + kc*32 + hi*8;
      #pragma unroll
      for (int j = 0; j < 4; ++j) {
        af[kc].u[j]   = *(const unsigned*)(arow + 2*j);
        af[kc].u[4+j] = *(const unsigned*)(arow + 16 + 2*j);
      }
    }
    // 8 N-tiles x 4 K-chunks = 32 WMMAs / wave / layer
    #pragma unroll
    for (int nt = 0; nt < 8; ++nt) {
      const int ncol = nt*16 + (lane & 15);
      #pragma unroll
      for (int kc = 0; kc < 4; ++kc) {
        Frag bf;
        const _Float16* brow = sWl + ncol*HID + kc*32 + hi*8;
        #pragma unroll
        for (int j = 0; j < 8; ++j) bf.u[j] = *(const unsigned*)(brow + 2*j + (j>>2)*8);
        acc[nt] = __builtin_amdgcn_wmma_f32_16x16x32_f16(
            false, af[kc].v, false, bf.v, (short)0, acc[nt], false, false);
      }
    }

    // ---- epilogue: bias + FiLM sin, write own rows back to sA
    const float* fql = fq + (layer+1)*HID;
    const float* phl = ph + (layer+1)*HID;
    const float* hbl = hid_b + layer*HID;
    #pragma unroll
    for (int nt = 0; nt < 8; ++nt) {
      const int ncol = nt*16 + (lane & 15);
      const float fqv = fql[ncol], phv = phl[ncol], hbv = hbl[ncol];
      #pragma unroll
      for (int r = 0; r < 8; ++r) {
        const int m = m0 + r + 8*hi;
        sA[m*HID + ncol] = (_Float16)__sinf(fqv*(acc[nt][r] + hbv) + phv);
      }
    }
    __syncthreads();   // all reads of sW[layer&1] done; sA complete

    // stream layer+2's weights into the buffer we just finished reading
    if (wave == 0 && (layer + 2) < (NLAYERS-1)) {
      tdm_load_tile_128x128_f16(hw16 + (size_t)(layer+2)*HID*HID, &sW[layer & 1][0]);
    }
  }

  // ---- head: 128 -> 4, sigmoid on rgb
  for (int j = 0; j < 2; ++j) {
    const int idx = j*256 + tid;            // 512 outputs
    const int m = idx >> 2, c = idx & 3;
    float v = out_b[c];
    for (int k = 0; k < HID; ++k) v += (float)sA[m*HID + k] * out_w[k*4 + c];
    if (c < 3) v = 1.f/(1.f + __expf(-v));
    out[(size_t)(base + m)*4 + c] = v;
  }
}

// ------------------------------------------- importance sampling per ray
__global__ void sample_fine_kernel(const float* __restrict__ cout, const float* __restrict__ zc,
                                   const float* __restrict__ tdirs, const float* __restrict__ torig,
                                   float* __restrict__ fz, float* __restrict__ fpts) {
  const int r = blockIdx.x*blockDim.x + threadIdx.x;
  if (r >= NRAYS) return;
  const int b = r / HW, pix = r % HW;
  const int pbase = b*PPB + pix*SSTEPS;

  float z[SSTEPS], w[SSTEPS];
  for (int i = 0; i < SSTEPS; ++i) z[i] = zc[pbase + i];
  float T = 1.f;
  for (int i = 0; i < SSTEPS; ++i) {
    const float delta = (i < SSTEPS-1) ? (z[i+1]-z[i]) : 1e10f;
    float sg = cout[(size_t)(pbase+i)*4 + 3];
    sg = sg > 0.f ? sg : 0.f;
    const float alpha = 1.f - __expf(-delta*sg);
    w[i] = alpha * T;
    T *= (1.f - alpha + 1e-10f);
  }
  float bins[SSTEPS-1], cdf[SSTEPS-1], pw[SSTEPS-2];
  for (int i = 0; i < SSTEPS-1; ++i) bins[i] = 0.5f*(z[i]+z[i+1]);
  float tot = 0.f;
  for (int i = 0; i < SSTEPS-2; ++i) { pw[i] = w[i+1] + 1e-5f; tot += pw[i]; }
  cdf[0] = 0.f;
  float c = 0.f;
  for (int i = 0; i < SSTEPS-2; ++i) { c += pw[i]/tot; cdf[i+1] = c; }

  const float ox = torig[r*3+0], oy = torig[r*3+1], oz = torig[r*3+2];
  const float dx = tdirs[r*3+0], dy = tdirs[r*3+1], dz = tdirs[r*3+2];
  for (int k = 0; k < SSTEPS; ++k) {
    const float u = u01hash((unsigned)(r*SSTEPS + k)*747796405u + 891u);
    int inds = 0;
    for (int j = 0; j < SSTEPS-1; ++j) inds += (cdf[j] <= u) ? 1 : 0;   // searchsorted right
    int below = inds - 1; below = below < 0 ? 0 : (below > SSTEPS-2 ? SSTEPS-2 : below);
    int above = inds;     above = above > SSTEPS-2 ? SSTEPS-2 : above;
    const float clo = cdf[below], chi = cdf[above];
    const float denom = (chi - clo < 1e-8f) ? 1.f : (chi - clo);
    const float fzv = bins[below] + (u - clo)/denom * (bins[above] - bins[below]);
    const int oidx = pbase + k;
    fz[oidx] = fzv;
    fpts[(size_t)oidx*3+0] = ox + dx*fzv;
    fpts[(size_t)oidx*3+1] = oy + dy*fzv;
    fpts[(size_t)oidx*3+2] = oz + dz*fzv;
  }
}

// ------------------------------------------------- merge + composite
__global__ void composite_kernel(const float* __restrict__ cout, const float* __restrict__ zc,
                                 const float* __restrict__ fout, const float* __restrict__ fz,
                                 float* __restrict__ out) {
  const int r = blockIdx.x*blockDim.x + threadIdx.x;
  if (r >= NRAYS) return;
  const int b = r / HW, pix = r % HW;
  const int pbase = b*PPB + pix*SSTEPS;
  const int N = 2*SSTEPS;
  float z[2*SSTEPS], cr[2*SSTEPS], cg[2*SSTEPS], cb[2*SSTEPS], sg[2*SSTEPS];
  for (int i = 0; i < SSTEPS; ++i) {            // fine first (concat order)
    const size_t q = (size_t)(pbase+i)*4;
    z[i]=fz[pbase+i]; cr[i]=fout[q]; cg[i]=fout[q+1]; cb[i]=fout[q+2]; sg[i]=fout[q+3];
  }
  for (int i = 0; i < SSTEPS; ++i) {
    const int j = SSTEPS + i;
    const size_t q = (size_t)(pbase+i)*4;
    z[j]=zc[pbase+i]; cr[j]=cout[q]; cg[j]=cout[q+1]; cb[j]=cout[q+2]; sg[j]=cout[q+3];
  }
  for (int i = 1; i < N; ++i) {                 // stable insertion sort by z
    const float kz=z[i],k0=cr[i],k1=cg[i],k2=cb[i],k3=sg[i];
    int j = i-1;
    while (j >= 0 && z[j] > kz) {
      z[j+1]=z[j]; cr[j+1]=cr[j]; cg[j+1]=cg[j]; cb[j+1]=cb[j]; sg[j+1]=sg[j]; --j;
    }
    z[j+1]=kz; cr[j+1]=k0; cg[j+1]=k1; cb[j+1]=k2; sg[j+1]=k3;
  }
  float T=1.f, R=0.f, G=0.f, Bl=0.f, D=0.f;
  for (int i = 0; i < N; ++i) {
    const float delta = (i < N-1) ? (z[i+1]-z[i]) : 1e10f;
    const float s = sg[i] > 0.f ? sg[i] : 0.f;
    const float alpha = 1.f - __expf(-delta*s);
    const float wt = alpha*T;
    T *= (1.f - alpha + 1e-10f);
    R += wt*cr[i]; G += wt*cg[i]; Bl += wt*cb[i]; D += wt*z[i];
  }
  out[b*3*HW + 0*HW + pix] = R *2.f - 1.f;
  out[b*3*HW + 1*HW + pix] = G *2.f - 1.f;
  out[b*3*HW + 2*HW + pix] = Bl*2.f - 1.f;
  const int py = pix / IMG, px = pix % IMG;
  const float x = -1.f + 2.f*(float)px/(float)(IMG-1);
  const float y =  1.f - 2.f*(float)py/(float)(IMG-1);
  const float zcam = -9.514364454222585f;
  const float inv = rsqrtf(x*x + y*y + zcam*zcam);
  out[BATCH*3*HW + b*HW + pix] = D * (-(zcam*inv));
}

// ---------------------------------------------------------------- launch
extern "C" void kernel_launch(void* const* d_in, const int* in_sizes, int n_in,
                              void* d_out, int out_size, void* d_ws, size_t ws_size,
                              hipStream_t stream) {
  (void)in_sizes; (void)n_in; (void)out_size; (void)ws_size;
  const float* z   = (const float*)d_in[0];
  const float* c2w = (const float*)d_in[1];
  const float* mw0 = (const float*)d_in[2];
  const float* mb0 = (const float*)d_in[3];
  const float* mw1 = (const float*)d_in[4];
  const float* mb1 = (const float*)d_in[5];
  const float* mw2 = (const float*)d_in[6];
  const float* mb2 = (const float*)d_in[7];
  const float* fw  = (const float*)d_in[8];
  const float* fb  = (const float*)d_in[9];
  const float* hw  = (const float*)d_in[10];
  const float* hb  = (const float*)d_in[11];
  const float* ow  = (const float*)d_in[12];
  const float* ob  = (const float*)d_in[13];
  float* out = (float*)d_out;

  float* ws = (float*)d_ws;
  size_t off = 0;
  auto alloc = [&](size_t n) { float* p = ws + off; off += (n + 63) & ~(size_t)63; return p; };
  float* freqs  = alloc((size_t)BATCH*FREQ_N);
  float* phases = alloc((size_t)BATCH*FREQ_N);
  float* tpts   = alloc((size_t)NPTS*3);
  float* zcrs   = alloc((size_t)NPTS);
  float* tdirs  = alloc((size_t)NRAYS*3);
  float* torig  = alloc((size_t)NRAYS*3);
  float* coutb  = alloc((size_t)NPTS*4);
  float* fzb    = alloc((size_t)NPTS);
  float* fpts   = alloc((size_t)NPTS*3);
  float* foutb  = alloc((size_t)NPTS*4);
  _Float16* hw16 = (_Float16*)alloc((size_t)(NLAYERS-1)*HID*HID/2);

  mapping_kernel<<<BATCH, 256, 0, stream>>>(z, mw0, mb0, mw1, mb1, mw2, mb2, freqs, phases);
  wcvt_kernel<<<((NLAYERS-1)*HID*HID + 255)/256, 256, 0, stream>>>(hw, hw16);
  gen_coarse_kernel<<<(NPTS + 255)/256, 256, 0, stream>>>(c2w, tpts, zcrs, tdirs, torig);
  siren_kernel<<<NPTS/HID, 256, 0, stream>>>(tpts, freqs, phases, fw, fb, hw16, hb, ow, ob, coutb);
  sample_fine_kernel<<<(NRAYS + 255)/256, 256, 0, stream>>>(coutb, zcrs, tdirs, torig, fzb, fpts);
  siren_kernel<<<NPTS/HID, 256, 0, stream>>>(fpts, freqs, phases, fw, fb, hw16, hb, ow, ob, foutb);
  composite_kernel<<<(NRAYS + 255)/256, 256, 0, stream>>>(coutb, zcrs, foutb, fzb, out);
}
